// EquivariantMessagePassing_11020886081992
// MI455X (gfx1250) — matrix operand order repaired and checked by reading, not compiled, not run on hardware
//
#include <hip/hip_runtime.h>
#include <hip/hip_bf16.h>
#include <math.h>

typedef __attribute__((ext_vector_type(16))) _Float16 v16h;
typedef __attribute__((ext_vector_type(4)))  _Float16 v4h;
typedef __attribute__((ext_vector_type(8)))  float    v8f;

#define N_NODES 10000
#define N_EDGES 320000
#define MUL 32
#define FEAT 288
#define N_INSTR 11

// INSTR: (0,0,0),(0,1,1),(0,2,2),(1,0,1),(1,1,0),(1,1,2),(1,2,1),(2,0,2),(2,1,1),(2,2,0),(2,2,2)

__device__ __forceinline__ float silu_f(float x) {
  // x * sigmoid(x) via hardware v_rcp_f32 (avoids the IEEE div expansion)
  return x * __builtin_amdgcn_rcpf(1.f + __expf(-x));
}

// ---------------------------------------------------------------------------
// Wigner-3j (real basis) computed exactly on device, mirroring the reference.
// ---------------------------------------------------------------------------
__device__ double dfact(int n) { double r = 1.0; for (int i = 2; i <= n; ++i) r *= (double)i; return r; }

__device__ double cg_coef(int l1, int m1, int l2, int m2, int l3, int m3) {
  if (m1 + m2 != m3) return 0.0;
  double pref = sqrt((double)(2 * l3 + 1) * dfact(l3 + l1 - l2) * dfact(l3 - l1 + l2) *
                     dfact(l1 + l2 - l3) / dfact(l1 + l2 + l3 + 1));
  pref *= sqrt(dfact(l3 + m3) * dfact(l3 - m3) * dfact(l1 - m1) * dfact(l1 + m1) *
               dfact(l2 - m2) * dfact(l2 + m2));
  double s = 0.0;
  for (int k = 0; k <= l1 + l2 - l3; ++k) {
    int d0 = k, d1 = l1 + l2 - l3 - k, d2 = l1 - m1 - k;
    int d3 = l2 + m2 - k, d4 = l3 - l2 + m1 + k, d5 = l3 - l1 - m2 + k;
    if (d0 < 0 || d1 < 0 || d2 < 0 || d3 < 0 || d4 < 0 || d5 < 0) continue;
    double den = dfact(d0) * dfact(d1) * dfact(d2) * dfact(d3) * dfact(d4) * dfact(d5);
    s += ((k & 1) ? -1.0 : 1.0) / den;
  }
  return pref * s;
}

__device__ void real_basis(int l, double Ar[5][5], double Ai[5][5]) {
  for (int r = 0; r < 5; ++r)
    for (int c = 0; c < 5; ++c) { Ar[r][c] = 0.0; Ai[r][c] = 0.0; }
  const double is2 = 0.70710678118654752440;
  Ar[l][l] = 1.0;
  for (int m = 1; m <= l; ++m) {
    double sgn = (m & 1) ? -1.0 : 1.0;
    Ar[l + m][l + m] = sgn * is2;
    Ar[l + m][l - m] = is2;
    Ai[l - m][l - m] = is2;
    Ai[l - m][l + m] = -sgn * is2;
  }
}

__global__ void w3j_kernel(float* __restrict__ out) {
  int t = threadIdx.x;
  if (t >= N_INSTR) return;
  const int L1[N_INSTR] = {0,0,0,1,1,1,1,2,2,2,2};
  const int L2[N_INSTR] = {0,1,2,0,1,1,2,0,1,2,2};
  const int L3[N_INSTR] = {0,1,2,1,0,2,1,2,1,0,2};
  int l1 = L1[t], l2 = L2[t], l3 = L3[t];
  int n1 = 2 * l1 + 1, n2 = 2 * l2 + 1, n3 = 2 * l3 + 1;

  double T[5][5][5];
  for (int a = 0; a < 5; ++a) for (int b = 0; b < 5; ++b) for (int c = 0; c < 5; ++c) T[a][b][c] = 0.0;
  for (int m1 = -l1; m1 <= l1; ++m1)
    for (int m2 = -l2; m2 <= l2; ++m2) {
      int m3 = -(m1 + m2);
      if (m3 < -l3 || m3 > l3) continue;
      int expo = l1 - l2 - m3;
      double sgn = ((expo % 2) != 0) ? -1.0 : 1.0;
      T[m1 + l1][m2 + l2][m3 + l3] =
          sgn / sqrt((double)(2 * l3 + 1)) * cg_coef(l1, m1, l2, m2, l3, m1 + m2);
    }

  double A1r[5][5], A1i[5][5], A2r[5][5], A2i[5][5], A3r[5][5], A3i[5][5];
  real_basis(l1, A1r, A1i);
  real_basis(l2, A2r, A2i);
  real_basis(l3, A3r, A3i);

  double Rr[5][5][5], Ri[5][5][5];
  double nr = 0.0, ni = 0.0;
  for (int i = 0; i < n1; ++i)
    for (int j = 0; j < n2; ++j)
      for (int k = 0; k < n3; ++k) {
        double rr = 0.0, ri = 0.0;
        for (int a = 0; a < n1; ++a)
          for (int b = 0; b < n2; ++b)
            for (int c = 0; c < n3; ++c) {
              double tv = T[a][b][c];
              if (tv == 0.0) continue;
              double p1r = A1r[i][a] * A2r[j][b] - A1i[i][a] * A2i[j][b];
              double p1i = A1r[i][a] * A2i[j][b] + A1i[i][a] * A2r[j][b];
              double pr = p1r * A3r[k][c] - p1i * A3i[k][c];
              double pi = p1r * A3i[k][c] + p1i * A3r[k][c];
              rr += pr * tv; ri += pi * tv;
            }
        Rr[i][j][k] = rr; Ri[i][j][k] = ri;
        nr += rr * rr; ni += ri * ri;
      }
  bool useR = (nr >= ni);
  double norm = sqrt(useR ? nr : ni);
  double inv = (norm > 0.0) ? 1.0 / norm : 0.0;
  for (int i = 0; i < 125; ++i) out[t * 125 + i] = 0.f;
  for (int i = 0; i < n1; ++i)
    for (int j = 0; j < n2; ++j)
      for (int k = 0; k < n3; ++k)
        out[t * 125 + i * 25 + j * 5 + k] =
            (float)((useR ? Rr[i][j][k] : Ri[i][j][k]) * inv);
}

// ---------------------------------------------------------------------------
// Weight conversion + swizzle into WMMA B-fragment order:
//   frag f, lane, q  ->  contiguous 16 halves per lane (one v16h load).
// B layout (ISA 7.12.2): element q holds K = (q&7) + 16*(q>>3) + 8*(lane>=16), N = lane&15.
// w1s: f = nt*2+ks (8 frags),  K = ks*32 + (q&7) + 16*(q>>3) + ((lane&16)>>1), N = nt*16+(lane&15)
// w2s: f = nt*2+ks (44 frags), same mapping, row stride 352
// lins: f = l*2+nt (6 frags),  u = (q&7) + 16*(q>>3) + ((lane&16)>>1), v = nt*16+(lane&15)
// ---------------------------------------------------------------------------
__global__ void convert_weights(const float* __restrict__ w1, const float* __restrict__ w2,
                                const float* __restrict__ la, const float* __restrict__ lb,
                                const float* __restrict__ lc,
                                _Float16* __restrict__ w1s, _Float16* __restrict__ w2s,
                                _Float16* __restrict__ lins) {
  const int NW1 = 8 * 512, NW2 = 44 * 512, NL = 6 * 512;
  int total = NW1 + NW2 + NL;
  for (int i = blockIdx.x * blockDim.x + threadIdx.x; i < total; i += gridDim.x * blockDim.x) {
    if (i < NW1) {
      int q = i & 15, lane = (i >> 4) & 31, f = i >> 9;
      int ks = f & 1, nt = f >> 1;
      int K = ks * 32 + (q & 7) + ((q >> 3) << 4) + ((lane & 16) >> 1);
      int N = nt * 16 + (lane & 15);
      w1s[i] = (_Float16)w1[K * 64 + N];
    } else if (i < NW1 + NW2) {
      int j = i - NW1;
      int q = j & 15, lane = (j >> 4) & 31, f = j >> 9;
      int ks = f & 1, nt = f >> 1;
      int K = ks * 32 + (q & 7) + ((q >> 3) << 4) + ((lane & 16) >> 1);
      int N = nt * 16 + (lane & 15);
      w2s[j] = (_Float16)w2[K * 352 + N];
    } else {
      int j = i - NW1 - NW2;
      int q = j & 15, lane = (j >> 4) & 31, f = j >> 9;
      int nt = f & 1, l = f >> 1;
      int u = (q & 7) + ((q >> 3) << 4) + ((lane & 16) >> 1);
      int v = nt * 16 + (lane & 15);
      const float* src = (l == 0) ? la : (l == 1) ? lb : lc;
      lins[j] = (_Float16)src[u * 32 + v];
    }
  }
}

// ---------------------------------------------------------------------------
// Tensor-product helper: fully unrolled per-instruction contraction
// ---------------------------------------------------------------------------
template <int T, int L1, int L2, int L3>
__device__ __forceinline__ void tp_one(const float* __restrict__ sW, float w,
                                       const float* x0, const float* x1, const float* x2,
                                       const float* sh, float* o0, float* o1, float* o2) {
  constexpr int D1 = 2 * L1 + 1, D2 = 2 * L2 + 1, D3 = 2 * L3 + 1;
  const float* xv = (L1 == 0) ? x0 : (L1 == 1) ? x1 : x2;
  const float* sp = sh + ((L2 == 0) ? 0 : (L2 == 1) ? 1 : 4);
  float* ov = (L3 == 0) ? o0 : (L3 == 1) ? o1 : o2;
  constexpr float alpha = (L3 == 0) ? 0.57735026919f : (L3 == 1) ? 0.86602540378f : 1.11803398875f;
  const float aw = alpha * w;
  const float* W = sW + T * 125;
#pragma unroll
  for (int k = 0; k < D3; ++k) {
    float s = 0.f;
#pragma unroll
    for (int i = 0; i < D1; ++i)
#pragma unroll
      for (int j = 0; j < D2; ++j)
        s += xv[i] * sp[j] * W[i * 25 + j * 5 + k];
    ov[k] += aw * s;
  }
}

// ---------------------------------------------------------------------------
// Fused edge kernel: SH + dist-MLP (WMMA) + tensor product + atomic scatter
// 4 waves/block, 16 edges/wave, 64 edges/block, 5000 blocks (exact).
// Per-wave LDS scratch s_buf aliases two phases (wave-private, DS in-order):
//   phase 1: h1 in A-fragment layout  [2][32][16] halves (1024)
//   phase 2: tpw transposed           [16 edge][32 u][12 instr-pad] (6144)
// ---------------------------------------------------------------------------
#define WPB 4
__global__ __launch_bounds__(128) void edge_kernel(
    const float* __restrict__ node_feat, const float* __restrict__ edge_vec,
    const int* __restrict__ edge_index, const float* __restrict__ fc_w0,
    const _Float16* __restrict__ w1s, const _Float16* __restrict__ w2s,
    const float* __restrict__ w3j, float* __restrict__ aggr) {
  __shared__ float s_w3j[N_INSTR * 125];
  __shared__ float s_sh[WPB][16][9];
  __shared__ int   s_src[WPB][16];
  __shared__ int   s_dst[WPB][16];
  __shared__ __align__(16) _Float16 s_buf[WPB][6144];

  const int lane = threadIdx.x & 31;
  const int wave = threadIdx.x >> 5;
  const int e0 = (blockIdx.x * WPB + wave) * 16;

  for (int i = threadIdx.x; i < N_INSTR * 125; i += blockDim.x) s_w3j[i] = w3j[i];

  // --- geometry + spherical harmonics (lane L handles edge e0 + (L&15)) ---
  const int m = lane & 15;
  const int e = e0 + m;
  float vx = edge_vec[3 * e + 0], vy = edge_vec[3 * e + 1], vz = edge_vec[3 * e + 2];
  float dist = sqrtf(vx * vx + vy * vy + vz * vz);
  float rinv = 1.f / fmaxf(dist, 1e-12f);
  float ux = vx * rinv, uy = vy * rinv, uz = vz * rinv;
  if (lane < 16) {
    s_sh[wave][m][0] = 1.f;
    s_sh[wave][m][1] = 1.73205080757f * uy;
    s_sh[wave][m][2] = 1.73205080757f * uz;
    s_sh[wave][m][3] = 1.73205080757f * ux;
    const float c = 3.87298334621f;  // sqrt(15)
    s_sh[wave][m][4] = c * ux * uy;
    s_sh[wave][m][5] = c * uy * uz;
    s_sh[wave][m][6] = 1.11803398875f * (3.f * uz * uz - 1.f);  // 0.5*sqrt(5)
    s_sh[wave][m][7] = c * ux * uz;
    s_sh[wave][m][8] = 0.5f * c * (ux * ux - uy * uy);
    int src = edge_index[e];
    s_src[wave][m] = src;
    s_dst[wave][m] = edge_index[N_EDGES + e];
    const char* row = (const char*)(node_feat + (size_t)src * FEAT);
#pragma unroll
    for (int o = 0; o < 1152; o += 256) __builtin_prefetch(row + o, 0, 0);
  }

  // --- layer 1: h0 = silu(dist * w0[k]); build A fragments (16x32 f16) ---
  // A layout: M = lane&15; K = (q&7) + 16*(q>>3) + 8*(lane>>4)
  v16h a0[2];
#pragma unroll
  for (int ks = 0; ks < 2; ++ks)
#pragma unroll
    for (int q = 0; q < 16; ++q) {
      int K = ks * 32 + (q & 7) + ((q >> 3) << 4) + ((lane >> 4) << 3);
      a0[ks][q] = (_Float16)silu_f(dist * fc_w0[K]);
    }

  // --- layer 2: h1 = silu((h0 @ W1) / 8), stored in A-fragment layout ---
#pragma unroll
  for (int nt = 0; nt < 4; ++nt) {
    v8f c = {};
#pragma unroll
    for (int ks = 0; ks < 2; ++ks) {
      v16h b = *(const v16h*)(w1s + (((size_t)(nt * 2 + ks) * 32 + lane) << 4));
      c = __builtin_amdgcn_wmma_f32_16x16x32_f16(false, a0[ks], false, b, (short)0, c, false, false);
    }
#pragma unroll
    for (int r = 0; r < 8; ++r) {  // C layout: M = r + 8*(lane>>4); N = nt*16+(lane&15)
      int M = r + ((lane >> 4) << 3);
      int N = nt * 16 + (lane & 15);
      // scatter into A-fragment layout for layer 3: K = N
      int ks2 = N >> 5, lk = N & 31;
      int g = (lk >> 3) & 1, hi = (lk >> 4) & 1;
      int q = (lk & 7) + (hi << 3);
      int lane2 = M + (g << 4);
      s_buf[wave][((ks2 * 32 + lane2) << 4) + q] = (_Float16)silu_f(c[r] * 0.125f);
    }
  }
  __syncthreads();

  // --- layer 3: tpw = (h1 @ W2) / 8 via WMMA ---
  v16h a1[2];
#pragma unroll
  for (int ks = 0; ks < 2; ++ks)
    a1[ks] = *(const v16h*)&s_buf[wave][(ks * 32 + lane) << 4];

#pragma unroll
  for (int nt = 0; nt < 22; ++nt) {
    v8f c = {};
#pragma unroll
    for (int ks = 0; ks < 2; ++ks) {
      v16h b = *(const v16h*)(w2s + (((size_t)(nt * 2 + ks) * 32 + lane) << 4));
      c = __builtin_amdgcn_wmma_f32_16x16x32_f16(false, a1[ks], false, b, (short)0, c, false, false);
    }
#pragma unroll
    for (int r = 0; r < 8; ++r) {
      int M = r + ((lane >> 4) << 3);
      int N = nt * 16 + (lane & 15);
      int t = N >> 5, uu = N & 31;     // instruction, multiplicity
      s_buf[wave][(M * 32 + uu) * 12 + t] = (_Float16)(c[r] * 0.125f);
    }
  }
  __syncthreads();

  // --- tensor product + scatter: lane = multiplicity u (MUL == wave32) ---
  const int u = lane;
  for (int el = 0; el < 16; ++el) {
    const float* nf = node_feat + (size_t)s_src[wave][el] * FEAT;
    float x0v[1], x1v[3], x2v[5], shv[9];
    x0v[0] = nf[u];
#pragma unroll
    for (int i = 0; i < 3; ++i) x1v[i] = nf[32 + u * 3 + i];
#pragma unroll
    for (int i = 0; i < 5; ++i) x2v[i] = nf[128 + u * 5 + i];
#pragma unroll
    for (int j = 0; j < 9; ++j) shv[j] = s_sh[wave][el][j];

    // vectorized read of the 11 per-instruction weights (12-half record)
    const v4h* tp = (const v4h*)&s_buf[wave][(el * 32 + u) * 12];
    v4h t0 = tp[0], t1 = tp[1], t2 = tp[2];
    float tw[11] = {(float)t0[0], (float)t0[1], (float)t0[2], (float)t0[3],
                    (float)t1[0], (float)t1[1], (float)t1[2], (float)t1[3],
                    (float)t2[0], (float)t2[1], (float)t2[2]};

    float o0[1] = {0.f}, o1[3] = {0.f, 0.f, 0.f}, o2[5] = {0.f, 0.f, 0.f, 0.f, 0.f};
    tp_one<0, 0, 0, 0>(s_w3j, tw[0], x0v, x1v, x2v, shv, o0, o1, o2);
    tp_one<1, 0, 1, 1>(s_w3j, tw[1], x0v, x1v, x2v, shv, o0, o1, o2);
    tp_one<2, 0, 2, 2>(s_w3j, tw[2], x0v, x1v, x2v, shv, o0, o1, o2);
    tp_one<3, 1, 0, 1>(s_w3j, tw[3], x0v, x1v, x2v, shv, o0, o1, o2);
    tp_one<4, 1, 1, 0>(s_w3j, tw[4], x0v, x1v, x2v, shv, o0, o1, o2);
    tp_one<5, 1, 1, 2>(s_w3j, tw[5], x0v, x1v, x2v, shv, o0, o1, o2);
    tp_one<6, 1, 2, 1>(s_w3j, tw[6], x0v, x1v, x2v, shv, o0, o1, o2);
    tp_one<7, 2, 0, 2>(s_w3j, tw[7], x0v, x1v, x2v, shv, o0, o1, o2);
    tp_one<8, 2, 1, 1>(s_w3j, tw[8], x0v, x1v, x2v, shv, o0, o1, o2);
    tp_one<9, 2, 2, 0>(s_w3j, tw[9], x0v, x1v, x2v, shv, o0, o1, o2);
    tp_one<10, 2, 2, 2>(s_w3j, tw[10], x0v, x1v, x2v, shv, o0, o1, o2);

    float* ag = aggr + (size_t)s_dst[wave][el] * FEAT;
    atomicAdd(&ag[u], o0[0]);
#pragma unroll
    for (int i = 0; i < 3; ++i) atomicAdd(&ag[32 + u * 3 + i], o1[i]);
#pragma unroll
    for (int i = 0; i < 5; ++i) atomicAdd(&ag[128 + u * 5 + i], o2[i]);
  }
}

// ---------------------------------------------------------------------------
// Node kernel: y = (1/sqrt(32)) * einsum('nui,uv->nvi') + node_feat, via WMMA
// Rows (n,i) tiled by 16, K = u = 32 (single WMMA k-step), N = v (2 tiles).
// Tiles: l0=625, l1=1875, l2=3125 (all exact); total 5625 wave-tiles.
// ---------------------------------------------------------------------------
__global__ __launch_bounds__(128) void node_kernel(const float* __restrict__ aggr,
                                                   const float* __restrict__ node_feat,
                                                   const _Float16* __restrict__ lins,
                                                   float* __restrict__ out) {
  const int lane = threadIdx.x & 31;
  const int wave = threadIdx.x >> 5;
  const int tile = blockIdx.x * 4 + wave;
  if (tile >= 5625) return;  // wave-uniform

  int l, rt;
  if (tile < 625)       { l = 0; rt = tile; }
  else if (tile < 2500) { l = 1; rt = tile - 625; }
  else                  { l = 2; rt = tile - 2500; }
  const int D = (l == 0) ? 1 : (l == 1) ? 3 : 5;
  const int START = (l == 0) ? 0 : (l == 1) ? 32 : 128;
  const int m0 = rt * 16;

  v16h a;
  {
    int mm = m0 + (lane & 15);
    int n = mm / D, i = mm % D;
    const float* arow = aggr + (size_t)n * FEAT + START + i;
#pragma unroll
    for (int q = 0; q < 16; ++q) {
      int u = (q & 7) + ((q >> 3) << 4) + ((lane >> 4) << 3);
      a[q] = (_Float16)arow[u * D];
    }
  }
#pragma unroll
  for (int nt = 0; nt < 2; ++nt) {
    v16h b = *(const v16h*)(lins + (((size_t)(l * 2 + nt) * 32 + lane) << 4));
    v8f c = {};
    c = __builtin_amdgcn_wmma_f32_16x16x32_f16(false, a, false, b, (short)0, c, false, false);
#pragma unroll
    for (int r = 0; r < 8; ++r) {
      int mm = m0 + r + ((lane >> 4) << 3);
      int n = mm / D, i = mm % D;
      int v = nt * 16 + (lane & 15);
      size_t idx = (size_t)n * FEAT + START + v * D + i;
      out[idx] = 0.17677669529f * c[r] + node_feat[idx];
    }
  }
}

// ---------------------------------------------------------------------------
extern "C" void kernel_launch(void* const* d_in, const int* in_sizes, int n_in,
                              void* d_out, int out_size, void* d_ws, size_t ws_size,
                              hipStream_t stream) {
  const float* node_feat = (const float*)d_in[0];
  const float* edge_vec  = (const float*)d_in[1];
  const int*   edge_index = (const int*)d_in[2];
  const float* fc_w0 = (const float*)d_in[3];
  const float* fc_w1 = (const float*)d_in[4];
  const float* fc_w2 = (const float*)d_in[5];
  const float* lin_w0 = (const float*)d_in[6];
  const float* lin_w1 = (const float*)d_in[7];
  const float* lin_w2 = (const float*)d_in[8];
  float* out = (float*)d_out;

  char* ws = (char*)d_ws;
  size_t off = 0;
  float* aggr = (float*)(ws + off); off += (size_t)N_NODES * FEAT * sizeof(float);
  float* w3j = (float*)(ws + off);  off += (size_t)N_INSTR * 125 * sizeof(float);
  off = (off + 255) & ~(size_t)255;
  _Float16* w1s  = (_Float16*)(ws + off); off += 8 * 512 * sizeof(_Float16);
  _Float16* w2s  = (_Float16*)(ws + off); off += 44 * 512 * sizeof(_Float16);
  _Float16* lins = (_Float16*)(ws + off); off += 6 * 512 * sizeof(_Float16);

  hipMemsetAsync(aggr, 0, (size_t)N_NODES * FEAT * sizeof(float), stream);
  w3j_kernel<<<1, 16, 0, stream>>>(w3j);
  convert_weights<<<64, 256, 0, stream>>>(fc_w1, fc_w2, lin_w0, lin_w1, lin_w2, w1s, w2s, lins);
  edge_kernel<<<N_EDGES / 64, 128, 0, stream>>>(node_feat, edge_vec, edge_index, fc_w0,
                                                w1s, w2s, w3j, aggr);
  node_kernel<<<(5625 + 3) / 4, 128, 0, stream>>>(aggr, node_feat, lins, out);
}
